// DGNNLayer_34084860461057
// MI455X (gfx1250) — compile-verified, matrix operand order checked by reading, not run on hardware
//
#include <hip/hip_runtime.h>
#include <hip/hip_bf16.h>
#include <math.h>

#define T_WIN   3
#define NNODES  50000
#define DIN     128
#define HIDDEN  128
#define NHEADS  8
#define DK      16
#define EPT     100000

typedef __attribute__((ext_vector_type(16))) _Float16 v16h;
typedef __attribute__((ext_vector_type(8)))  _Float16 v8h;
typedef __attribute__((ext_vector_type(8)))  float    v8f;
typedef unsigned int __attribute__((ext_vector_type(4))) uint4v;
typedef int          __attribute__((ext_vector_type(4))) int4v;
typedef int          __attribute__((ext_vector_type(8))) int8v;

__device__ __forceinline__ float gelu_exact(float v) {
    return 0.5f * v * (1.0f + erff(v * 0.7071067811865476f));
}

__device__ __forceinline__ void atomicMaxF(float* addr, float val) {
    int* ai = (int*)addr;
    int cur = __float_as_int(*addr);
    while (__int_as_float(cur) < val) {
        int assumed = cur;
        cur = atomicCAS(ai, assumed, __float_as_int(val));
        if (cur == assumed) break;
    }
}
__device__ __forceinline__ void atomicMinF(float* addr, float val) {
    int* ai = (int*)addr;
    int cur = __float_as_int(*addr);
    while (__int_as_float(cur) > val) {
        int assumed = cur;
        cur = atomicCAS(ai, assumed, __float_as_int(val));
        if (cur == assumed) break;
    }
}

// ---------------------------------------------------------------------------
// f32 -> f16 elementwise copy (one-time activation conversion)
// ---------------------------------------------------------------------------
__global__ void f32_to_f16_kernel(const float* __restrict__ src,
                                  _Float16* __restrict__ dst, size_t n)
{
    size_t i = (size_t)blockIdx.x * blockDim.x + threadIdx.x;
    if (i < n) dst[i] = (_Float16)src[i];
}

// ---------------------------------------------------------------------------
// Pack a f32 row-major weight W[K x Nout] into f16 WMMA B-fragments:
// frag[(kt*(Nout/16)+nt)][lane][0..15] laid out per ISA 16-bit B 32x16 layout.
// One wave per (kt, nt). One-time cost; makes GEMM B loads contiguous 32B/lane.
// ---------------------------------------------------------------------------
__global__ __launch_bounds__(32)
void pack_weight_kernel(const float* __restrict__ W, int ldw, int Nout,
                        _Float16* __restrict__ Wfrag)
{
    const int lane   = threadIdx.x & 31;
    const int halfid = lane >> 4;
    const int nc     = lane & 15;
    const int koff   = halfid * 8;
    const int k0     = blockIdx.x * 32;
    const int n      = blockIdx.y * 16 + nc;
    v16h b;
    const float* wcol = W + (size_t)k0 * ldw + n;
#pragma unroll
    for (int i = 0; i < 8; ++i) {
        b[i]     = (_Float16)wcol[(size_t)(koff + i) * ldw];
        b[i + 8] = (_Float16)wcol[(size_t)(16 + koff + i) * ldw];
    }
    *(v16h*)(Wfrag + ((size_t)(blockIdx.x * (Nout >> 4) + blockIdx.y) * 32 + lane) * 16) = b;
}

// ---------------------------------------------------------------------------
// WMMA GEMM, one wave -> 16x64 output tile, 4 accumulators.
// A16: row-major f16 [M x K]; whole 16xK A-slab staged into LDS by the
// Tensor Data Mover (tensor_load_to_lds + s_wait_tensorcnt), then consumed
// as ds_load_b128 fragments. B comes from pre-packed fragments (32B/lane).
// epilogue: 0 = f32 store, 1 = GELU -> f16 store, 2 = +resid -> f32 store
// ---------------------------------------------------------------------------
template<int K>
__global__ __launch_bounds__(32)
void wmma_gemm_tdm_kernel(const _Float16* __restrict__ A16,
                          const _Float16* __restrict__ Wfrag,
                          const float* __restrict__ bias,
                          float* __restrict__ Cf,
                          _Float16* __restrict__ Ch,
                          int ldc, int Nout,
                          int epilogue,
                          const float* __restrict__ resid)
{
    __shared__ _Float16 smA[16 * 256];          // up to 8 KB (K<=256)

    const int lane   = threadIdx.x & 31;
    const int halfid = lane >> 4;
    const int mr     = lane & 15;
    const size_t row0 = (size_t)blockIdx.x * 16;
    const int    col0 = blockIdx.y * 64;
    const int    koff = halfid * 8;

    // ---- TDM: DMA the 16 x K f16 A-slab into LDS (D# per ISA ch.8) ----
    {
        unsigned long long gaddr = (unsigned long long)(const void*)(A16 + row0 * K);
        unsigned lds_off = (unsigned)(uintptr_t)&smA[0];
        uint4v g0; int8v g1; int4v gz4 = {0, 0, 0, 0};
        int8v  gz8 = {0, 0, 0, 0, 0, 0, 0, 0};
        g0[0] = 1u;                                              // count=1, no gather
        g0[1] = lds_off;                                         // lds_addr
        g0[2] = (unsigned)(gaddr & 0xFFFFFFFFull);               // global_addr[31:0]
        g0[3] = (unsigned)((gaddr >> 32) & 0x1FFFFFFull)         // global_addr[56:32]
              | (2u << 30);                                      // type = 2 (image)
        const unsigned td0 = (unsigned)K;     // tensor_dim0 (elements)
        const unsigned td1 = 16u;             // tensor_dim1 (rows of this tile)
        const unsigned ti0 = (unsigned)K;     // tile_dim0
        const unsigned ti1 = 16u;             // tile_dim1
        const unsigned long long s0 = (unsigned long long)K;     // dim0 stride
        g1[0] = (int)(1u << 16);                                 // data_size = 2 bytes
        g1[1] = (int)((td0 & 0xFFFFu) << 16);                    // tensor_dim0 lo
        g1[2] = (int)((td0 >> 16) | ((td1 & 0xFFFFu) << 16));    // dim0 hi | dim1 lo
        g1[3] = (int)((td1 >> 16) | ((ti0 & 0xFFFFu) << 16));    // dim1 hi | tile_dim0
        g1[4] = (int)(ti1 & 0xFFFFu);                            // tile_dim1 (tile_dim2=0)
        g1[5] = (int)(s0 & 0xFFFFFFFFull);                       // stride0[31:0]
        g1[6] = (int)((s0 >> 32) & 0xFFFFu);                     // stride0[47:32]
        g1[7] = 0;
        __builtin_amdgcn_tensor_load_to_lds(g0, g1, gz4, gz4, gz8, 0);
        __builtin_amdgcn_s_wait_tensorcnt(0);
    }

    v8f acc[4] = {};
#pragma unroll
    for (int k0 = 0; k0 < K; k0 += 32) {
        // A fragment from LDS: two ds_load_b128 per lane
        v16h a;
        const _Float16* ap = &smA[mr * K + k0 + koff];
        v8h alo = *(const v8h*)ap;
        v8h ahi = *(const v8h*)(ap + 16);
#pragma unroll
        for (int i = 0; i < 8; ++i) { a[i] = alo[i]; a[i + 8] = ahi[i]; }

        const int kt = k0 >> 5;
#pragma unroll
        for (int nt = 0; nt < 4; ++nt) {
            const _Float16* bp = Wfrag +
                ((size_t)(kt * (Nout >> 4) + (col0 >> 4) + nt) * 32 + lane) * 16;
            v16h b = *(const v16h*)bp;               // contiguous 32B/lane
            acc[nt] = __builtin_amdgcn_wmma_f32_16x16x32_f16(
                false, a, false, b, (short)0, acc[nt], false, false);
        }
    }

#pragma unroll
    for (int nt = 0; nt < 4; ++nt) {
        const int   n  = col0 + nt * 16 + mr;
        const float bn = bias ? bias[n] : 0.0f;
#pragma unroll
        for (int r = 0; r < 8; ++r) {
            size_t m = row0 + (size_t)(8 * halfid + r);
            float vv = acc[nt][r] + bn;
            if (epilogue == 1)      Ch[m * (size_t)ldc + n] = (_Float16)gelu_exact(vv);
            else if (epilogue == 2) Cf[m * (size_t)ldc + n] = vv + resid[m * (size_t)ldc + n];
            else                    Cf[m * (size_t)ldc + n] = vv;
        }
    }
}

// ---------------------------------------------------------------------------
// Per-t_tar initialization of softmax stats and message accumulators
// ---------------------------------------------------------------------------
__global__ void init_stats_kernel(float* __restrict__ amax, float* __restrict__ amin,
                                  float* __restrict__ sumc, float* __restrict__ sums,
                                  float* __restrict__ hatc, float* __restrict__ hats)
{
    size_t i = (size_t)blockIdx.x * blockDim.x + threadIdx.x;
    const size_t nstats = (size_t)NNODES * NHEADS;
    const size_t nhid   = (size_t)NNODES * HIDDEN;
    if (i < nstats) {
        amax[i] = -INFINITY; amin[i] = INFINITY;
        sumc[i] = 0.0f;      sums[i] = 0.0f;
    }
    if (i < nhid) { hatc[i] = 0.0f; hats[i] = 0.0f; }
}

// Pass 1: att[e,h] = (q[tar]·k[src])/sqrt(16); per-(node,head) max & min
__global__ void edge_scores_kernel(const float* __restrict__ q, const float* __restrict__ k,
                                   const int* __restrict__ src, const int* __restrict__ tar,
                                   float* __restrict__ att,
                                   float* __restrict__ amax, float* __restrict__ amin)
{
    int gid = blockIdx.x * blockDim.x + threadIdx.x;
    int e = gid >> 3, h = gid & 7;
    if (e >= EPT) return;
    int s = src[e], t = tar[e];
    const float* qp = q + (size_t)t * HIDDEN + h * DK;
    const float* kp = k + (size_t)s * HIDDEN + h * DK;
    float a = 0.0f;
#pragma unroll
    for (int d = 0; d < DK; ++d) a += qp[d] * kp[d];
    a *= 0.25f;
    att[(size_t)e * NHEADS + h] = a;
    atomicMaxF(&amax[t * NHEADS + h], a);
    atomicMinF(&amin[t * NHEADS + h], a);
}

// Pass 2: exp sums for causal (+a) and spurious (-a) softmaxes
__global__ void edge_expsum_kernel(const float* __restrict__ att, const int* __restrict__ tar,
                                   const float* __restrict__ amax, const float* __restrict__ amin,
                                   float* __restrict__ sumc, float* __restrict__ sums)
{
    int gid = blockIdx.x * blockDim.x + threadIdx.x;
    int e = gid >> 3, h = gid & 7;
    if (e >= EPT) return;
    int t = tar[e];
    float a = att[(size_t)e * NHEADS + h];
    atomicAdd(&sumc[t * NHEADS + h], __expf(a - amax[t * NHEADS + h]));
    atomicAdd(&sums[t * NHEADS + h], __expf(amin[t * NHEADS + h] - a));
}

// Pass 3: scatter p*v (one thread per edge-channel)
__global__ void edge_scatter_kernel(const float* __restrict__ att, const float* __restrict__ v,
                                    const int* __restrict__ src, const int* __restrict__ tar,
                                    const float* __restrict__ amax, const float* __restrict__ amin,
                                    const float* __restrict__ sumc, const float* __restrict__ sums,
                                    float* __restrict__ hatc, float* __restrict__ hats)
{
    size_t gid = (size_t)blockIdx.x * blockDim.x + threadIdx.x;
    int e = (int)(gid >> 7);
    int c = (int)(gid & 127);
    if (e >= EPT) return;
    int h = c >> 4;
    int s = src[e], t = tar[e];
    float a  = att[(size_t)e * NHEADS + h];
    float pc = __expf(a - amax[t * NHEADS + h]) / (sumc[t * NHEADS + h] + 1e-16f);
    float ps = __expf(amin[t * NHEADS + h] - a) / (sums[t * NHEADS + h] + 1e-16f);
    float vv = v[(size_t)s * HIDDEN + c];
    atomicAdd(&hatc[(size_t)t * HIDDEN + c], pc * vv);
    atomicAdd(&hats[(size_t)t * HIDDEN + c], ps * vv);
}

// Optional x-residual add (in place) + LayerNorm; writes normalized f16 rows
__global__ __launch_bounds__(HIDDEN)
void add_ln_kernel(const float* __restrict__ xt, float* __restrict__ hat,
                   _Float16* __restrict__ hn16,
                   const float* __restrict__ lns, const float* __restrict__ lnb,
                   int addx)
{
    const int node = blockIdx.x;
    const int c    = threadIdx.x;
    float* hp = hat + (size_t)node * HIDDEN;
    float hv = hp[c];
    if (addx) { hv += xt[(size_t)node * HIDDEN + c]; hp[c] = hv; }
    __shared__ float red[HIDDEN];
    red[c] = hv; __syncthreads();
    for (int off = 64; off > 0; off >>= 1) { if (c < off) red[c] += red[c + off]; __syncthreads(); }
    float mu = red[0] * (1.0f / HIDDEN); __syncthreads();
    float d = hv - mu;
    red[c] = d * d; __syncthreads();
    for (int off = 64; off > 0; off >>= 1) { if (c < off) red[c] += red[c + off]; __syncthreads(); }
    float var = red[0] * (1.0f / HIDDEN);
    hn16[(size_t)node * HIDDEN + c] =
        (_Float16)(d * rsqrtf(var + 1e-5f) * lns[c] + lnb[c]);
}

__global__ void combine_kernel(float* __restrict__ ox,
                               const float* __restrict__ oc, const float* __restrict__ os)
{
    size_t i = (size_t)blockIdx.x * blockDim.x + threadIdx.x;
    if (i < (size_t)NNODES * HIDDEN) ox[i] = oc[i] + os[i];
}

// ---------------------------------------------------------------------------
static inline char* carve(char*& p, size_t bytes) {
    char* r = p;
    p += (bytes + 255) & ~(size_t)255;
    return r;
}

extern "C" void kernel_launch(void* const* d_in, const int* in_sizes, int n_in,
                              void* d_out, int out_size, void* d_ws, size_t ws_size,
                              hipStream_t stream)
{
    (void)in_sizes; (void)n_in; (void)out_size; (void)ws_size;

    const float* x   = (const float*)d_in[0];
    const int*   ei  = (const int*)d_in[1];
    const float* Wq  = (const float*)d_in[2];
    const float* bq  = (const float*)d_in[3];
    const float* Wk  = (const float*)d_in[4];
    const float* bk  = (const float*)d_in[5];
    const float* Wv  = (const float*)d_in[6];
    const float* bv  = (const float*)d_in[7];
    const float* lns = (const float*)d_in[8];
    const float* lnb = (const float*)d_in[9];
    const float* W1  = (const float*)d_in[10];
    const float* b1  = (const float*)d_in[11];
    const float* W2  = (const float*)d_in[12];
    const float* b2  = (const float*)d_in[13];
    float* out = (float*)d_out;

    const size_t NH = (size_t)NNODES * HIDDEN;
    char* wp = (char*)d_ws;
    float*    q_all = (float*)carve(wp, T_WIN * NH * 4);
    float*    k_all = (float*)carve(wp, T_WIN * NH * 4);
    float*    v_all = (float*)carve(wp, T_WIN * NH * 4);
    float*    att   = (float*)carve(wp, (size_t)3 * EPT * NHEADS * 4);
    float*    amax  = (float*)carve(wp, (size_t)NNODES * NHEADS * 4);
    float*    amin  = (float*)carve(wp, (size_t)NNODES * NHEADS * 4);
    float*    sumc  = (float*)carve(wp, (size_t)NNODES * NHEADS * 4);
    float*    sums  = (float*)carve(wp, (size_t)NNODES * NHEADS * 4);
    float*    hatc  = (float*)carve(wp, NH * 4);
    float*    hats  = (float*)carve(wp, NH * 4);
    _Float16* x16   = (_Float16*)carve(wp, T_WIN * NH * 2);
    _Float16* hn16  = (_Float16*)carve(wp, NH * 2);
    _Float16* act16 = (_Float16*)carve(wp, (size_t)NNODES * 256 * 2);
    _Float16* wfQ   = (_Float16*)carve(wp, (size_t)DIN * HIDDEN * 2);
    _Float16* wfK   = (_Float16*)carve(wp, (size_t)DIN * HIDDEN * 2);
    _Float16* wfV   = (_Float16*)carve(wp, (size_t)DIN * HIDDEN * 2);
    _Float16* wf1   = (_Float16*)carve(wp, (size_t)HIDDEN * 2 * HIDDEN * 2);
    _Float16* wf2   = (_Float16*)carve(wp, (size_t)2 * HIDDEN * HIDDEN * 2);

    // ---- one-time conversions / weight packing ----
    {
        size_t n = T_WIN * NH;
        f32_to_f16_kernel<<<(unsigned)((n + 255) / 256), 256, 0, stream>>>(x, x16, n);
        pack_weight_kernel<<<dim3(DIN / 32, HIDDEN / 16), 32, 0, stream>>>(Wq, HIDDEN, HIDDEN, wfQ);
        pack_weight_kernel<<<dim3(DIN / 32, HIDDEN / 16), 32, 0, stream>>>(Wk, HIDDEN, HIDDEN, wfK);
        pack_weight_kernel<<<dim3(DIN / 32, HIDDEN / 16), 32, 0, stream>>>(Wv, HIDDEN, HIDDEN, wfV);
        pack_weight_kernel<<<dim3(HIDDEN / 32, (2 * HIDDEN) / 16), 32, 0, stream>>>(W1, 2 * HIDDEN, 2 * HIDDEN, wf1);
        pack_weight_kernel<<<dim3((2 * HIDDEN) / 32, HIDDEN / 16), 32, 0, stream>>>(W2, HIDDEN, HIDDEN, wf2);
    }

    // ---- QKV projections: [150000 x 128] @ [128 x 128], WMMA + TDM ----
    {
        dim3 g((T_WIN * NNODES) / 16, HIDDEN / 64);
        wmma_gemm_tdm_kernel<DIN><<<g, 32, 0, stream>>>(x16, wfQ, bq, q_all, nullptr, HIDDEN, HIDDEN, 0, nullptr);
        wmma_gemm_tdm_kernel<DIN><<<g, 32, 0, stream>>>(x16, wfK, bk, k_all, nullptr, HIDDEN, HIDDEN, 0, nullptr);
        wmma_gemm_tdm_kernel<DIN><<<g, 32, 0, stream>>>(x16, wfV, bv, v_all, nullptr, HIDDEN, HIDDEN, 0, nullptr);
    }

    const int edge_blocks    = (EPT * NHEADS) / 256;   // 3125
    const int scatter_blocks = (EPT * HIDDEN) / 256;   // 50000
    const int init_blocks    = (int)(NH / 256);        // 25000

    for (int t = 0; t < T_WIN; ++t) {
        const float* qt = q_all + (size_t)t * NH;
        const float* xt = x     + (size_t)t * NH;

        init_stats_kernel<<<init_blocks, 256, 0, stream>>>(amax, amin, sumc, sums, hatc, hats);

        for (int ts = 0; ts <= t; ++ts) {
            const int* srcp = ei + (size_t)ts * 2 * EPT;
            const int* tarp = srcp + EPT;
            edge_scores_kernel<<<edge_blocks, 256, 0, stream>>>(
                qt, k_all + (size_t)ts * NH, srcp, tarp,
                att + (size_t)ts * EPT * NHEADS, amax, amin);
        }
        for (int ts = 0; ts <= t; ++ts) {
            const int* srcp = ei + (size_t)ts * 2 * EPT;
            const int* tarp = srcp + EPT;
            edge_expsum_kernel<<<edge_blocks, 256, 0, stream>>>(
                att + (size_t)ts * EPT * NHEADS, tarp, amax, amin, sumc, sums);
        }
        for (int ts = 0; ts <= t; ++ts) {
            const int* srcp = ei + (size_t)ts * 2 * EPT;
            const int* tarp = srcp + EPT;
            edge_scatter_kernel<<<scatter_blocks, 256, 0, stream>>>(
                att + (size_t)ts * EPT * NHEADS, v_all + (size_t)ts * NH,
                srcp, tarp, amax, amin, sumc, sums, hatc, hats);
        }

        // ---- FFN for causal (s=0, +x residual input) and spurious (s=1) ----
        for (int s = 0; s < 2; ++s) {
            float* hat  = (s == 0) ? hatc : hats;
            float* outp = out + ((size_t)(1 + s) * T_WIN + t) * NH;

            add_ln_kernel<<<NNODES, HIDDEN, 0, stream>>>(xt, hat, hn16, lns, lnb, (s == 0) ? 1 : 0);

            // act16 = gelu(hn @ W1 + b1), f16 store  [N,128]x[128,256]
            wmma_gemm_tdm_kernel<HIDDEN><<<dim3(NNODES / 16, (2 * HIDDEN) / 64), 32, 0, stream>>>(
                hn16, wf1, b1, nullptr, act16, 2 * HIDDEN, 2 * HIDDEN, 1, nullptr);

            // outp = hat + act @ W2 + b2   [N,256]x[256,128]
            wmma_gemm_tdm_kernel<2 * HIDDEN><<<dim3(NNODES / 16, HIDDEN / 64), 32, 0, stream>>>(
                act16, wf2, b2, outp, nullptr, HIDDEN, HIDDEN, 2, hat);
        }

        combine_kernel<<<init_blocks, 256, 0, stream>>>(
            out + ((size_t)0 * T_WIN + t) * NH,
            out + ((size_t)1 * T_WIN + t) * NH,
            out + ((size_t)2 * T_WIN + t) * NH);
    }
}
